// VectorQuantizer_69148973465935
// MI455X (gfx1250) — compile-verified, matrix operand order checked by reading, not run on hardware
//
#include <hip/hip_runtime.h>
#include <stdint.h>

#define B_ROWS  32768
#define K_CODES 8192
#define D_DIM   512
#define BETA_F  0.25f

#define BM 128          // z_e rows per workgroup (8 waves x 16 rows)
#define BN 64           // codebook rows per K-slab (double buffered in LDS)
#define LS 520          // LDS row stride in bf16 elems: 260 dwords -> 4-bank shift/row,
                        // 16 rows x 4-dword b128 reads cover all 64 banks conflict-free
#define LSB (LS * 2)    // row stride in bytes (1040, 16B aligned)

typedef __attribute__((ext_vector_type(16))) __bf16 v16bf;
typedef __attribute__((ext_vector_type(8)))  __bf16 v8bf;
typedef __attribute__((ext_vector_type(8)))  float  v8f;

union Frag { v16bf v; v8bf h[2]; };

__device__ __forceinline__ unsigned short f32_to_bf16(float f) {
    unsigned int u = __float_as_uint(f);
    u += 0x7FFFu + ((u >> 16) & 1u);          // round-to-nearest-even
    return (unsigned short)(u >> 16);
}

// ---------------------------------------------------------------------------
// Kernel 1: embed fp32 -> bf16 copy (packed) + e_sq[k] = sum_d embed[k,d]^2
// ---------------------------------------------------------------------------
__global__ __launch_bounds__(256) void vq_prep(
    const float* __restrict__ embed,
    unsigned int* __restrict__ embBfPacked,   // K*D/2 uints (2 bf16 per uint)
    float* __restrict__ eSq) {
    __shared__ float red[256];
    const int k = blockIdx.x;
    const int t = threadIdx.x;
    const float* row = embed + (size_t)k * D_DIM;
    float a = row[2 * t];
    float b = row[2 * t + 1];
    embBfPacked[(size_t)k * (D_DIM / 2) + t] =
        (unsigned int)f32_to_bf16(a) | ((unsigned int)f32_to_bf16(b) << 16);
    red[t] = a * a + b * b;
    __syncthreads();
    for (int s = 128; s > 0; s >>= 1) {
        if (t < s) red[t] += red[t + s];
        __syncthreads();
    }
    if (t == 0) eSq[k] = red[0];
}

// ---------------------------------------------------------------------------
// Async stage of one BN x D bf16 slab into LDS via GLOBAL_LOAD_ASYNC_TO_LDS.
// 256 threads, 64 rows x 1024B: thread -> (row = tid/4, quarter = tid%4),
// each thread issues 16 b128 async copies (ASYNCcnt max 16 per wave).
// ---------------------------------------------------------------------------
__device__ __forceinline__ void issue_slab_async(
    const unsigned short* __restrict__ embBf, unsigned ldsBase, int nBase, int tid) {
    const int row = tid >> 2;
    const int q   = tid & 3;
    const unsigned long long gbase = (unsigned long long)(uintptr_t)embBf;
    unsigned goff = (unsigned)((nBase + row) * (D_DIM * 2) + q * 256);
    unsigned lds  = ldsBase + (unsigned)(row * LSB + q * 256);
    #pragma unroll
    for (int i = 0; i < 16; ++i) {
        asm volatile("global_load_async_to_lds_b128 %0, %1, %2"
                     :: "v"(lds), "v"(goff), "s"(gbase) : "memory");
        lds  += 16;
        goff += 16;
    }
}

// ---------------------------------------------------------------------------
// Kernel 2: argmin_k ( e_sq[k] - 2 * <z_e[b], embed[k]> )  via bf16 WMMA,
// with double-buffered async LDS staging of the codebook.
// grid = B_ROWS / BM blocks x 256 threads (8 waves, wave32)
// ---------------------------------------------------------------------------
__global__ __launch_bounds__(256) void vq_argmin(
    const float* __restrict__ zE,
    const unsigned short* __restrict__ embBf,  // bf16 bit patterns, row-major K x D
    const float* __restrict__ eSq,
    int* __restrict__ idxOut,
    float* __restrict__ idxOutF) {
    __shared__ unsigned short zeL[BM * LS];        // 130 KB
    __shared__ unsigned short emL[2 * BN * LS];    // 2 x 65 KB double buffer

    const int tid  = threadIdx.x;
    const int wave = tid >> 5;
    const int lane = tid & 31;
    const int l16  = lane & 15;
    const int hw   = lane >> 4;                    // half-wave select
    const int rowBase = blockIdx.x * BM;

    const unsigned emBase  = (unsigned)(size_t)(void*)&emL[0];
    const unsigned bufStep = (unsigned)(BN * LSB);

    // Prefetch slab 0 asynchronously, then stage z_e (overlapped with it).
    issue_slab_async(embBf, emBase, 0, tid);

    {   // z_e rows -> LDS as bf16 (each thread: half a row)
        const int r = tid >> 1, seg = tid & 1;
        const float4* src =
            (const float4*)(zE + (size_t)(rowBase + r) * D_DIM + seg * 256);
        unsigned short* dst = zeL + r * LS + seg * 256;
        #pragma unroll 4
        for (int i = 0; i < 64; ++i) {
            float4 f = src[i];
            unsigned int u0 = (unsigned int)f32_to_bf16(f.x) |
                              ((unsigned int)f32_to_bf16(f.y) << 16);
            unsigned int u1 = (unsigned int)f32_to_bf16(f.z) |
                              ((unsigned int)f32_to_bf16(f.w) << 16);
            uint2 u; u.x = u0; u.y = u1;
            *(uint2*)(dst + 4 * i) = u;
        }
    }

    float best[8];
    int   bidx[8];
    #pragma unroll
    for (int v = 0; v < 8; ++v) { best[v] = 3.4e38f; bidx[v] = 0; }

    // A/B-fragment element offsets within a 32-wide k-chunk (ISA 16-bit layout):
    // lanes 0-15: VGPR0-3 -> k 0..7,  VGPR4-7 -> k 16..23
    // lanes16-31: VGPR0-3 -> k 8..15, VGPR4-7 -> k 24..31
    const int offLo = 8 * hw;
    const int offHi = 16 + 8 * hw;
    const unsigned short* zRow = zeL + (wave * 16 + l16) * LS;

    const int nSlabs = K_CODES / BN;   // 128
    for (int s = 0; s < nSlabs; ++s) {
        const int nBase = s * BN;
        const unsigned short* emBuf = emL + (unsigned)(s & 1) * (BN * LS);

        // Wait for this wave's async copies of slab s, then rendezvous:
        // after the barrier every wave's slab-s data is in LDS and buffer
        // (s+1)%2 is free (its consumers finished in iteration s-1).
        asm volatile("s_wait_asynccnt 0x0" ::: "memory");
        __syncthreads();
        if (s + 1 < nSlabs)
            issue_slab_async(embBf, emBase + (unsigned)((s + 1) & 1) * bufStep,
                             nBase + BN, tid);

        v8f acc[4] = {v8f{}, v8f{}, v8f{}, v8f{}};
        const unsigned short* bp[4];
        #pragma unroll
        for (int t4 = 0; t4 < 4; ++t4)
            bp[t4] = emBuf + (t4 * 16 + l16) * LS;

        #pragma unroll 4
        for (int c = 0; c < D_DIM / 32; ++c) {
            const int o0 = 32 * c + offLo;
            const int o1 = 32 * c + offHi;
            Frag a;
            a.h[0] = *(const v8bf*)(zRow + o0);
            a.h[1] = *(const v8bf*)(zRow + o1);
            #pragma unroll
            for (int t4 = 0; t4 < 4; ++t4) {
                Frag fb;
                fb.h[0] = *(const v8bf*)(bp[t4] + o0);
                fb.h[1] = *(const v8bf*)(bp[t4] + o1);
                acc[t4] = __builtin_amdgcn_wmma_f32_16x16x32_bf16(
                    false, a.v, false, fb.v, (short)0, acc[t4], false, false);
            }
        }

        // score = e_sq[n] - 2*dot ; running argmin in C-matrix layout
        // (lane = column, VGPR v = row v (+8 for upper half-wave))
        #pragma unroll
        for (int t4 = 0; t4 < 4; ++t4) {
            const int col = nBase + t4 * 16 + l16;
            const float es = eSq[col];
            #pragma unroll
            for (int v = 0; v < 8; ++v) {
                float sc = fmaf(-2.0f, acc[t4][v], es);
                if (sc < best[v]) { best[v] = sc; bidx[v] = col; }
            }
        }
    }

    // Cross-lane min+argmin over the 16 lanes sharing each row
    // (masks < 16 keep the two half-wave row groups independent on wave32)
    #pragma unroll
    for (int v = 0; v < 8; ++v) {
        float bv = best[v];
        int   bi = bidx[v];
        #pragma unroll
        for (int m = 1; m < 16; m <<= 1) {
            float ov = __shfl_xor(bv, m, 32);
            int   oi = __shfl_xor(bi, m, 32);
            if (ov < bv || (ov == bv && oi < bi)) { bv = ov; bi = oi; }
        }
        if (l16 == 0) {
            const int grow = rowBase + wave * 16 + v + 8 * hw;
            idxOut[grow]  = bi;
            idxOutF[grow] = (float)bi;
        }
    }
}

// ---------------------------------------------------------------------------
// Kernel 3: z_q_st = embed[idx] (exact fp32 gather) + partial sum (z-q)^2
// ---------------------------------------------------------------------------
__global__ __launch_bounds__(256) void vq_gather(
    const float* __restrict__ zE,
    const float* __restrict__ embed,
    const int* __restrict__ idx,
    float* __restrict__ zqOut,
    float* __restrict__ partial) {
    __shared__ float red[256];
    const int t = threadIdx.x;
    const int rBase = blockIdx.x * 16;
    float s = 0.0f;
    for (int e = t; e < 16 * D_DIM; e += 256) {
        const int r = e >> 9;
        const int d = e & (D_DIM - 1);
        const int grow = rBase + r;
        const int k = idx[grow];
        const float q = embed[(size_t)k * D_DIM + d];
        const float z = zE[(size_t)grow * D_DIM + d];
        zqOut[(size_t)grow * D_DIM + d] = q;    // z_e + sg(z_q - z_e) == z_q
        const float df = z - q;
        s += df * df;
    }
    red[t] = s;
    __syncthreads();
    for (int st = 128; st > 0; st >>= 1) {
        if (t < st) red[t] += red[t + st];
        __syncthreads();
    }
    if (t == 0) partial[blockIdx.x] = red[0];
}

// ---------------------------------------------------------------------------
// Kernel 4: vq_loss = (1+beta) * mean((z_e - z_q)^2)
// ---------------------------------------------------------------------------
__global__ __launch_bounds__(256) void vq_loss(
    const float* __restrict__ partial, int n, float* __restrict__ out) {
    __shared__ float red[256];
    const int t = threadIdx.x;
    float s = 0.0f;
    for (int i = t; i < n; i += 256) s += partial[i];
    red[t] = s;
    __syncthreads();
    for (int st = 128; st > 0; st >>= 1) {
        if (t < st) red[t] += red[t + st];
        __syncthreads();
    }
    if (t == 0)
        out[0] = red[0] * (1.0f + BETA_F) / ((float)B_ROWS * (float)D_DIM);
}

// ---------------------------------------------------------------------------
extern "C" void kernel_launch(void* const* d_in, const int* in_sizes, int n_in,
                              void* d_out, int out_size, void* d_ws, size_t ws_size,
                              hipStream_t stream) {
    (void)in_sizes; (void)n_in; (void)out_size; (void)ws_size;
    const float* zE    = (const float*)d_in[0];   // [B, D]
    const float* embed = (const float*)d_in[1];   // [K, D]

    // workspace layout
    char* ws = (char*)d_ws;
    unsigned short* embBf = (unsigned short*)(ws);                 // K*D*2  = 8,388,608 B
    float* eSq            = (float*)(ws + 8388608);                // K*4    =    32,768 B
    int*   idxBuf         = (int*)  (ws + 8421376);                // B*4    =   131,072 B
    float* partial        = (float*)(ws + 8552448);                // 2048*4 =     8,192 B

    float* out     = (float*)d_out;
    float* zqOut   = out;                                   // [B*D]
    float* idxOutF = out + (size_t)B_ROWS * D_DIM;          // [B] (cast to float)
    float* lossOut = idxOutF + B_ROWS;                      // [1]

    vq_prep<<<K_CODES, 256, 0, stream>>>(embed, (unsigned int*)embBf, eSq);
    vq_argmin<<<B_ROWS / BM, 256, 0, stream>>>(zE, embBf, eSq, idxBuf, idxOutF);
    vq_gather<<<B_ROWS / 16, 256, 0, stream>>>(zE, embed, idxBuf, zqOut, partial);
    vq_loss<<<1, 256, 0, stream>>>(partial, B_ROWS / 16, lossOut);
}
